// TransMIL_65644280152872
// MI455X (gfx1250) — matrix operand order repaired and checked by reading, not compiled, not run on hardware
//
#include <hip/hip_runtime.h>

// ---------------------------------------------------------------------------
// TransMIL forward for gfx1250 (MI455X).
// All GEMMs: v_wmma_f32_16x16x32_bf16, wave32, one 16x64 output strip per wave
// (1 A-frag reused over 4 B-frags -> 4 WMMA per K-step).  B operands are kept
// in [N][K] (transposed) bf16 storage so every fragment load is contiguous
// 16x bf16 (vectorizes to global_load_b128 pairs).
// ---------------------------------------------------------------------------

typedef __attribute__((ext_vector_type(16))) __bf16 bf16x16;
typedef __attribute__((ext_vector_type(8)))  float  f32x8;

#define NTOK 16130   // cls + 16129 feature tokens
#define NPAD 16384   // nystrom padded length
#define PADR 254     // zero-pad rows at front
#define CDIM 512
#define QKVD 1536
#define NHEAD 8
#define DHD 64
#define MLM 256      // landmarks
#define N0 16000
#define K0 1024
#define GRD 127      // 127x127 spatial grid for ppeg

// ---------------- helpers ----------------

__device__ inline __bf16 f2bf(float f) {
  union { float f; unsigned u; } x; x.f = f;
  unsigned r = x.u + 0x7FFFu + ((x.u >> 16) & 1u);
  unsigned short h = (unsigned short)(r >> 16);
  __bf16 o; __builtin_memcpy(&o, &h, 2);
  return o;
}

__device__ inline float wred_sum(float v) {
#pragma unroll
  for (int o = 16; o >= 1; o >>= 1) v += __shfl_xor(v, o, 32);
  return v;
}
__device__ inline float wred_max(float v) {
#pragma unroll
  for (int o = 16; o >= 1; o >>= 1) v = fmaxf(v, __shfl_xor(v, o, 32));
  return v;
}

// ---------------- batched WMMA GEMM (A row-major, B transposed) -------------
// C[M,N] = epilogue(A[M,K] @ B[K,N]) with B stored as Bt[N][K].
// mode 0: v = alpha*acc + bias (+relu);  mode 1: v = identv*I - acc.
// Optional bf16 mirrors: Cbf row-major [M][N], CbfT transposed [N][M]
// (so the result can directly feed the A or B side of a later GEMM).
// Each wave computes a 16x64 strip: 4 accumulators, 4 WMMA per K-step.

__global__ __launch_bounds__(128)
void gemm_bf16_tn(const __bf16* __restrict__ A, int lda, long sA,
                  const __bf16* __restrict__ Bt, int ldb, long sB,
                  float* __restrict__ C, int ldc, long sC,
                  __bf16* __restrict__ Cbf, int ldcbf, long sCbf,
                  __bf16* __restrict__ CbfT, int ldcbt, long sCbt,
                  const float* __restrict__ bias,
                  int M, int N, int K,
                  float alpha, float identv, int mode, int relu)
{
  int batch = blockIdx.z;
  A  += (long)batch * sA;
  Bt += (long)batch * sB;
  C  += (long)batch * sC;
  if (Cbf)  Cbf  += (long)batch * sCbf;
  if (CbfT) CbfT += (long)batch * sCbt;

  int wave = threadIdx.x >> 5;
  int lane = threadIdx.x & 31;
  int ngrp = N >> 6;                 // 64-wide column groups
  int grp = blockIdx.x * 4 + wave;
  if (grp >= ngrp) return;
  int col0 = grp << 6;
  int row  = (int)blockIdx.y << 4;
  int mr   = lane & 15;
  int hi   = (lane >> 4) & 1;

  f32x8 acc[4];
#pragma unroll
  for (int t = 0; t < 4; ++t) acc[t] = (f32x8){0.f,0.f,0.f,0.f,0.f,0.f,0.f,0.f};

  // A 16x32 bf16 frag: lanes 0-15 hold K {0..15},
  // lanes 16-31 hold K {16..31}; row m = lane%16.
  const __bf16* pa  = A  + (long)(row + mr) * lda + hi * 16;
  // B 32x16 frag from Bt[N][K]: lanes 0-15 K kk..kk+15, lanes 16-31
  // K kk+16..kk+31; column n = lane%16 -> 16 contiguous bf16 per lane.
  const __bf16* pb0 = Bt + (long)(col0 + mr) * ldb + hi * 16;

  for (int kk = 0; kk < K; kk += 32) {
    bf16x16 a;
#pragma unroll
    for (int r = 0; r < 16; ++r) { a[r] = pa[kk + r]; }
    __builtin_prefetch(pa + kk + 128, 0, 1);
#pragma unroll
    for (int t = 0; t < 4; ++t) {
      const __bf16* pb = pb0 + (long)t * 16 * ldb + kk;
      bf16x16 b;
#pragma unroll
      for (int j = 0; j < 16; ++j) b[j] = pb[j];
      __builtin_prefetch(pb + 128, 0, 1);
      acc[t] = __builtin_amdgcn_wmma_f32_16x16x32_bf16(false, a, false, b,
                                                       (short)0, acc[t], false, false);
    }
  }

  int n = lane & 15;
  int mbase = row + hi * 8;
#pragma unroll
  for (int t = 0; t < 4; ++t) {
    int col = col0 + (t << 4);
    float bv = bias ? bias[col + n] : 0.f;
#pragma unroll
    for (int i = 0; i < 8; ++i) {
      int mm = mbase + i;
      float v;
      if (mode == 1) {
        v = identv * ((mm == (col + n)) ? 1.f : 0.f) - acc[t][i];
      } else {
        v = alpha * acc[t][i] + bv;
        if (relu) v = fmaxf(v, 0.f);
      }
      C[(long)mm * ldc + col + n] = v;
      if (Cbf)  Cbf[(long)mm * ldcbf + col + n] = f2bf(v);
      if (CbfT) CbfT[(long)(col + n) * ldcbt + mm] = f2bf(v);
    }
  }
}

// ---------------- elementwise / reduction kernels ----------------

__global__ void f32_to_bf16_scale(const float* __restrict__ in,
                                  __bf16* __restrict__ out, long n, float scale)
{
  long i = blockIdx.x * (long)blockDim.x + threadIdx.x;
  long st = (long)gridDim.x * blockDim.x;
  for (; i < n; i += st) out[i] = f2bf(in[i] * scale);
}

// out[c][r] = bf16(in[r][c])  -- weight transpose-convert to [N][K]
__global__ void transpose_to_bf16(const float* __restrict__ in,
                                  __bf16* __restrict__ out, int R, int Cc)
{
  long idx = blockIdx.x * (long)blockDim.x + threadIdx.x;
  long st = (long)gridDim.x * blockDim.x;
  long total = (long)R * Cc;
  for (; idx < total; idx += st) {
    long r = idx / Cc;
    long c = idx - r * Cc;
    out[c * (long)R + r] = f2bf(in[idx]);
  }
}

// vT[h][d][t] = bf16(QKV[t][1024 + h*64 + d])  -- v transposed per head
__global__ void v_transpose_bf(const float* __restrict__ QKV, __bf16* __restrict__ vT)
{
  long idx = blockIdx.x * (long)blockDim.x + threadIdx.x;
  long st = (long)gridDim.x * blockDim.x;
  long total = (long)NHEAD * DHD * NPAD;
  for (; idx < total; idx += st) {
    int t = (int)(idx & (NPAD - 1));
    int d = (int)((idx >> 14) & 63);
    int h = (int)(idx >> 20);
    vT[idx] = f2bf(QKV[(long)t * QKVD + 1024 + h * DHD + d]);
  }
}

__global__ void zero_bf(__bf16* __restrict__ p, long n)
{
  long i = blockIdx.x * (long)blockDim.x + threadIdx.x;
  long st = (long)gridDim.x * blockDim.x;
  for (; i < n; i += st) p[i] = f2bf(0.f);
}

__global__ void build_T(const float* __restrict__ H, const float* __restrict__ cls,
                        float* __restrict__ T)
{
  long i = blockIdx.x * (long)blockDim.x + threadIdx.x;
  long st = (long)gridDim.x * blockDim.x;
  long total = (long)NTOK * CDIM;
  for (; i < total; i += st) {
    long t = i >> 9; int c = (int)(i & 511);
    if (t == 0) T[i] = cls[c];
    else {
      long s = t - 1;
      if (s >= N0) s -= N0;        // wraparound duplication of first rows
      T[i] = H[s * CDIM + c];
    }
  }
}

// LayerNorm of ntok tokens -> bf16 at row offset rowOff; one wave per token.
__global__ __launch_bounds__(128)
void layernorm_to_bf(const float* __restrict__ X, const float* __restrict__ g,
                     const float* __restrict__ b, __bf16* __restrict__ out,
                     long rowOff, int ntok)
{
  int tok = blockIdx.x * 4 + (threadIdx.x >> 5);
  if (tok >= ntok) return;
  int lane = threadIdx.x & 31;
  const float* p = X + (long)tok * CDIM;
  float s = 0.f, s2 = 0.f;
  for (int i = lane; i < CDIM; i += 32) { float v = p[i]; s += v; s2 += v * v; }
  s = wred_sum(s); s2 = wred_sum(s2);
  float mu = s / (float)CDIM;
  float var = s2 / (float)CDIM - mu * mu;
  float inv = rsqrtf(var + 1e-5f);
  __bf16* o = out + (rowOff + tok) * CDIM;
  for (int i = lane; i < CDIM; i += 32)
    o[i] = f2bf((p[i] - mu) * inv * g[i] + b[i]);
}

__global__ void scale_q(float* __restrict__ QKV)
{
  long i = blockIdx.x * (long)blockDim.x + threadIdx.x;
  long st = (long)gridDim.x * blockDim.x;
  long total = (long)NPAD * CDIM;
  for (; i < total; i += st) {
    long r = i >> 9; int c = (int)(i & 511);
    QKV[r * QKVD + c] *= 0.125f;      // DH^-0.5
  }
}

// ql/kl landmark means: mean over 64 consecutive tokens per landmark.
__global__ void landmarks(const float* __restrict__ QKV,
                          float* __restrict__ qlf, float* __restrict__ klf,
                          __bf16* __restrict__ qlb, __bf16* __restrict__ klb)
{
  long idx = blockIdx.x * (long)blockDim.x + threadIdx.x;
  long total = (long)NHEAD * MLM * DHD;
  long st = (long)gridDim.x * blockDim.x;
  for (; idx < total; idx += st) {
    int d = (int)(idx & 63);
    int m = (int)((idx >> 6) & 255);
    int h = (int)(idx >> 14);
    const float* q = QKV + (long)(m * 64) * QKVD + h * DHD + d;
    const float* k = q + CDIM;
    float sq = 0.f, sk = 0.f;
    for (int j = 0; j < 64; ++j) { sq += q[(long)j * QKVD]; sk += k[(long)j * QKVD]; }
    sq *= (1.f / 64.f); sk *= (1.f / 64.f);
    long o = ((long)h * MLM + m) * DHD + d;
    qlf[o] = sq; klf[o] = sk;
    qlb[o] = f2bf(sq); klb[o] = f2bf(sk);
  }
}

// softmax over rows of width 256; one wave per row; optional f32 + bf16 out.
__global__ __launch_bounds__(256)
void softmax256(const float* __restrict__ in, float* __restrict__ outf,
                __bf16* __restrict__ outb, long rows)
{
  long row = (long)blockIdx.x * 8 + (threadIdx.x >> 5);
  if (row >= rows) return;
  int lane = threadIdx.x & 31;
  const float* p = in + row * 256;
  float v[8], mx = -3.4e38f;
#pragma unroll
  for (int i = 0; i < 8; ++i) { v[i] = p[lane + i * 32]; mx = fmaxf(mx, v[i]); }
  mx = wred_max(mx);
  float s = 0.f;
#pragma unroll
  for (int i = 0; i < 8; ++i) { v[i] = __expf(v[i] - mx); s += v[i]; }
  s = wred_sum(s);
  float inv = 1.f / s;
#pragma unroll
  for (int i = 0; i < 8; ++i) {
    float r = v[i] * inv;
    long o = row * 256 + lane + i * 32;
    if (outf) outf[o] = r;
    if (outb) outb[o] = f2bf(r);
  }
}

// softmax over rows of width `width` (16384); one block per row.
__global__ __launch_bounds__(256)
void softmax_wide(const float* __restrict__ in, __bf16* __restrict__ outb, int width)
{
  __shared__ float sbuf[8];
  __shared__ float sval;
  long row = blockIdx.x;
  const float* p = in + row * (long)width;
  int tid = threadIdx.x, lane = tid & 31, wid = tid >> 5;

  float mx = -3.4e38f;
  for (int i = tid; i < width; i += 256) mx = fmaxf(mx, p[i]);
  mx = wred_max(mx);
  if (lane == 0) sbuf[wid] = mx;
  __syncthreads();
  if (tid == 0) {
    float m = sbuf[0];
    for (int i = 1; i < 8; ++i) m = fmaxf(m, sbuf[i]);
    sval = m;
  }
  __syncthreads();
  mx = sval;
  __syncthreads();

  float s = 0.f;
  for (int i = tid; i < width; i += 256) s += __expf(p[i] - mx);
  s = wred_sum(s);
  if (lane == 0) sbuf[wid] = s;
  __syncthreads();
  if (tid == 0) {
    float t = 0.f;
    for (int i = 0; i < 8; ++i) t += sbuf[i];
    sval = t;
  }
  __syncthreads();
  float inv = 1.f / sval;

  __bf16* o = outb + row * (long)width;
  for (int i = tid; i < width; i += 256) o[i] = f2bf(__expf(p[i] - mx) * inv);
}

// per-head max |row sum| and max |col sum| of a2 (256x256)
__global__ __launch_bounds__(256)
void pinv_stats(const float* __restrict__ a2, float* __restrict__ stats)
{
  __shared__ float buf[256];
  int h = blockIdx.x;
  const float* x = a2 + (long)h * 65536;
  int t = threadIdx.x;
  float rs = 0.f, cs = 0.f;
  for (int j = 0; j < 256; ++j) {
    rs += fabsf(x[t * 256 + j]);
    cs += fabsf(x[j * 256 + t]);
  }
  buf[t] = rs; __syncthreads();
  for (int s = 128; s > 0; s >>= 1) { if (t < s) buf[t] = fmaxf(buf[t], buf[t + s]); __syncthreads(); }
  if (t == 0) stats[h] = buf[0];
  __syncthreads();
  buf[t] = cs; __syncthreads();
  for (int s = 128; s > 0; s >>= 1) { if (t < s) buf[t] = fmaxf(buf[t], buf[t + s]); __syncthreads(); }
  if (t == 0) stats[8 + h] = buf[0];
}

// z0 = a2^T / (max_col * max_row)  (global max over all heads, per reference)
// writes z f32 row-major, zb bf16 row-major (A-side), zbT bf16 [N][K] (B-side)
__global__ void pinv_make_z(const float* __restrict__ a2,
                            const float* __restrict__ stats,
                            float* __restrict__ z, __bf16* __restrict__ zb,
                            __bf16* __restrict__ zbT)
{
  float mr = 0.f, mc = 0.f;
  for (int i = 0; i < 8; ++i) { mr = fmaxf(mr, stats[i]); mc = fmaxf(mc, stats[8 + i]); }
  float s = 1.f / (mr * mc);
  long idx = blockIdx.x * (long)blockDim.x + threadIdx.x;
  long st = (long)gridDim.x * blockDim.x;
  long total = (long)NHEAD * 65536;
  for (; idx < total; idx += st) {
    long h = idx >> 16;
    long e = idx & 65535;
    int i = (int)(e >> 8), j = (int)(e & 255);
    float v = a2[h * 65536 + (long)j * 256 + i] * s;   // z[i][j] = x[j][i]*s
    z[idx] = v;
    zb[idx] = f2bf(v);
    zbT[idx] = f2bf(a2[h * 65536 + (long)i * 256 + j] * s);  // zT[i][j] = z[j][i]
  }
}

// t1T = (7*I - xz)^T  (bf16, B-side layout)
__global__ void seven_i_minus_T(const float* __restrict__ xz, __bf16* __restrict__ outT)
{
  long idx = blockIdx.x * (long)blockDim.x + threadIdx.x;
  long st = (long)gridDim.x * blockDim.x;
  long total = (long)NHEAD * 65536;
  for (; idx < total; idx += st) {
    long h = idx >> 16;
    long e = idx & 65535;
    int jn = (int)(e >> 8), ik = (int)(e & 255);
    // outT[jn][ik] = t1[ik][jn] = 7*delta - xz[ik][jn]
    outT[idx] = f2bf(((jn == ik) ? 7.f : 0.f) - xz[h * 65536 + (long)ik * 256 + jn]);
  }
}

// depthwise residual conv (kernel 33 along the sequence), added into O.
__global__ void res_conv(float* __restrict__ O, const float* __restrict__ QKV,
                         const float* __restrict__ w)
{
  long idx = blockIdx.x * (long)blockDim.x + threadIdx.x;
  long st = (long)gridDim.x * blockDim.x;
  long total = (long)NPAD * CDIM;
  for (; idx < total; idx += st) {
    int c = (int)(idx & 511);
    long n = idx >> 9;
    int h = c >> 6;
    const float* wv = w + h * 33;
    float s = 0.f;
    for (int t = 0; t < 33; ++t) {
      long r = n + t - 16;
      if (r >= 0 && r < NPAD) s += QKV[r * QKVD + 1024 + c] * wv[t];
    }
    O[n * CDIM + c] += s;
  }
}

// T[t] += Y[254 + t]   (drop front padding)
__global__ void residual_add(float* __restrict__ T, const float* __restrict__ Y)
{
  long idx = blockIdx.x * (long)blockDim.x + threadIdx.x;
  long st = (long)gridDim.x * blockDim.x;
  long total = (long)NTOK * CDIM;
  for (; idx < total; idx += st) {
    long t = idx >> 9; int c = (int)(idx & 511);
    T[idx] += Y[(PADR + t) * CDIM + c];
  }
}

// PPEG: identity + depthwise 7x7 + 5x5 + 3x3 on the 127x127 token grid.
__global__ void ppeg_kernel(const float* __restrict__ Tin, float* __restrict__ Tout,
                            const float* __restrict__ w7, const float* __restrict__ b7,
                            const float* __restrict__ w5, const float* __restrict__ b5,
                            const float* __restrict__ w3, const float* __restrict__ b3)
{
  long idx = blockIdx.x * (long)blockDim.x + threadIdx.x;
  long st = (long)gridDim.x * blockDim.x;
  long total = (long)NTOK * CDIM;
  for (; idx < total; idx += st) {
    long t = idx >> 9; int ch = (int)(idx & 511);
    if (t == 0) { Tout[idx] = Tin[idx]; continue; }
    long n = t - 1;
    int r = (int)(n / GRD), cc = (int)(n % GRD);
    auto val = [&](int rr, int c2) -> float {
      if (rr < 0 || rr >= GRD || c2 < 0 || c2 >= GRD) return 0.f;
      return Tin[(1 + (long)rr * GRD + c2) * CDIM + ch];
    };
    float acc = val(r, cc) + b7[ch] + b5[ch] + b3[ch];
    const float* W7 = w7 + ch * 49;
    for (int dy = -3; dy <= 3; ++dy)
      for (int dx = -3; dx <= 3; ++dx)
        acc += val(r + dy, cc + dx) * W7[(dy + 3) * 7 + (dx + 3)];
    const float* W5 = w5 + ch * 25;
    for (int dy = -2; dy <= 2; ++dy)
      for (int dx = -2; dx <= 2; ++dx)
        acc += val(r + dy, cc + dx) * W5[(dy + 2) * 5 + (dx + 2)];
    const float* W3 = w3 + ch * 9;
    for (int dy = -1; dy <= 1; ++dy)
      for (int dx = -1; dx <= 1; ++dx)
        acc += val(r + dy, cc + dx) * W3[(dy + 1) * 3 + (dx + 1)];
    Tout[idx] = acc;
  }
}

// final LN of token 0 + fc2 + softmax + argmax -> 5 floats
__global__ __launch_bounds__(256)
void head_kernel(const float* __restrict__ T, const float* __restrict__ g,
                 const float* __restrict__ b, const float* __restrict__ w2,
                 const float* __restrict__ b2, float* __restrict__ out)
{
  __shared__ float xs[CDIM];
  __shared__ float rs[8], rs2[8];
  __shared__ float smu, sinv;
  int tid = threadIdx.x, lane = tid & 31, wid = tid >> 5;
  float s = 0.f, s2 = 0.f;
  for (int i = tid; i < CDIM; i += 256) { float v = T[i]; s += v; s2 += v * v; }
  s = wred_sum(s); s2 = wred_sum(s2);
  if (lane == 0) { rs[wid] = s; rs2[wid] = s2; }
  __syncthreads();
  if (tid == 0) {
    float a = 0.f, c = 0.f;
    for (int i = 0; i < 8; ++i) { a += rs[i]; c += rs2[i]; }
    float mu = a / (float)CDIM;
    float var = c / (float)CDIM - mu * mu;
    smu = mu; sinv = rsqrtf(var + 1e-5f);
  }
  __syncthreads();
  for (int i = tid; i < CDIM; i += 256) xs[i] = (T[i] - smu) * sinv * g[i] + b[i];
  __syncthreads();
  if (tid == 0) {
    float l0 = b2[0], l1 = b2[1];
    for (int i = 0; i < CDIM; ++i) { l0 += xs[i] * w2[i * 2]; l1 += xs[i] * w2[i * 2 + 1]; }
    float m = fmaxf(l0, l1);
    float e0 = __expf(l0 - m), e1 = __expf(l1 - m);
    float inv = 1.f / (e0 + e1);
    out[0] = l0; out[1] = l1;
    out[2] = e0 * inv; out[3] = e1 * inv;
    out[4] = (l1 > l0) ? 1.f : 0.f;
  }
}

// ---------------- host orchestration ----------------

extern "C" void kernel_launch(void* const* d_in, const int* in_sizes, int n_in,
                              void* d_out, int out_size, void* d_ws, size_t ws_size,
                              hipStream_t stream)
{
  (void)in_sizes; (void)n_in; (void)out_size; (void)ws_size;

  const float* data   = (const float*)d_in[0];
  const float* fc1_w  = (const float*)d_in[1];
  const float* fc1_b  = (const float*)d_in[2];
  const float* cls    = (const float*)d_in[3];
  const float* l1_ng  = (const float*)d_in[4];
  const float* l1_nb  = (const float*)d_in[5];
  const float* l1_qkv = (const float*)d_in[6];
  const float* l1_ow  = (const float*)d_in[7];
  const float* l1_ob  = (const float*)d_in[8];
  const float* l1_rw  = (const float*)d_in[9];
  const float* pw7 = (const float*)d_in[10]; const float* pb7 = (const float*)d_in[11];
  const float* pw5 = (const float*)d_in[12]; const float* pb5 = (const float*)d_in[13];
  const float* pw3 = (const float*)d_in[14]; const float* pb3 = (const float*)d_in[15];
  const float* l2_ng  = (const float*)d_in[16];
  const float* l2_nb  = (const float*)d_in[17];
  const float* l2_qkv = (const float*)d_in[18];
  const float* l2_ow  = (const float*)d_in[19];
  const float* l2_ob  = (const float*)d_in[20];
  const float* l2_rw  = (const float*)d_in[21];
  const float* nrm_g  = (const float*)d_in[22];
  const float* nrm_b  = (const float*)d_in[23];
  const float* fc2_w  = (const float*)d_in[24];
  const float* fc2_b  = (const float*)d_in[25];

  // ---- workspace bump allocator ----
  char* base = (char*)d_ws;
  size_t off = 0;
  auto alloc = [&](size_t bytes) -> void* {
    void* p = base + off;
    off = (off + bytes + 255) & ~(size_t)255;
    return p;
  };

  float*  T    = (float*)alloc((size_t)NTOK * CDIM * 4);
  float*  T2b  = (float*)alloc((size_t)NTOK * CDIM * 4);
  __bf16* Wbf  = (__bf16*)alloc((size_t)1024 * 1536 * 2);   // current weight^T, bf16
  __bf16* R1   = (__bf16*)alloc((size_t)NPAD * QKVD * 2);   // data_bf / qkv_bf
  float*  R2   = (float*)alloc((size_t)NPAD * QKVD * 4);    // Hbuf / QKV f32 / Y
  __bf16* XPB  = (__bf16*)alloc((size_t)NPAD * CDIM * 2);   // padded LN(x), bf16
  __bf16* VTB  = (__bf16*)alloc((size_t)NHEAD * DHD * NPAD * 2); // v^T per head
  float*  QLF  = (float*)alloc((size_t)NHEAD * MLM * DHD * 4);
  float*  KLF  = (float*)alloc((size_t)NHEAD * MLM * DHD * 4);
  __bf16* QLB  = (__bf16*)alloc((size_t)NHEAD * MLM * DHD * 2);
  __bf16* KLB  = (__bf16*)alloc((size_t)NHEAD * MLM * DHD * 2);
  float*  BIG  = (float*)alloc((size_t)NHEAD * NPAD * MLM * 4); // a1 then a3 logits
  __bf16* A1B  = (__bf16*)alloc((size_t)NHEAD * NPAD * MLM * 2);
  __bf16* A3B  = (__bf16*)alloc((size_t)NHEAD * MLM * NPAD * 2);
  float*  A2F  = (float*)alloc((size_t)NHEAD * 65536 * 4);
  __bf16* A2B  = (__bf16*)alloc((size_t)NHEAD * 65536 * 2);
  float*  ZF   = (float*)alloc((size_t)NHEAD * 65536 * 4);
  __bf16* ZB   = (__bf16*)alloc((size_t)NHEAD * 65536 * 2);
  __bf16* ZBT  = (__bf16*)alloc((size_t)NHEAD * 65536 * 2);
  float*  Z2F  = (float*)alloc((size_t)NHEAD * 65536 * 4);
  __bf16* Z2B  = (__bf16*)alloc((size_t)NHEAD * 65536 * 2);
  __bf16* Z2BT = (__bf16*)alloc((size_t)NHEAD * 65536 * 2);
  float*  XZF  = (float*)alloc((size_t)NHEAD * 65536 * 4);
  __bf16* XZB  = (__bf16*)alloc((size_t)NHEAD * 65536 * 2);
  __bf16* T1BT = (__bf16*)alloc((size_t)NHEAD * 65536 * 2);
  float*  U2F  = (float*)alloc((size_t)NHEAD * 65536 * 4);
  __bf16* U2BT = (__bf16*)alloc((size_t)NHEAD * 65536 * 2);
  float*  U3F  = (float*)alloc((size_t)NHEAD * 65536 * 4);
  __bf16* U3BT = (__bf16*)alloc((size_t)NHEAD * 65536 * 2);
  float*  G1F  = (float*)alloc((size_t)NHEAD * MLM * DHD * 4);
  __bf16* G1BT = (__bf16*)alloc((size_t)NHEAD * MLM * DHD * 2); // [64][256] per head
  float*  G2F  = (float*)alloc((size_t)NHEAD * MLM * DHD * 4);
  __bf16* G2BT = (__bf16*)alloc((size_t)NHEAD * MLM * DHD * 2); // [64][256] per head
  float*  OF   = (float*)alloc((size_t)NPAD * CDIM * 4);
  __bf16* OB   = (__bf16*)alloc((size_t)NPAD * CDIM * 2);
  float*  STAT = (float*)alloc(16 * 4);

  auto eg = [](long n) { return dim3((unsigned)((n + 255) / 256)); };

  // A [M][K] row-major bf16, Bt [N][K] bf16.  Optional Cbf [M][N] / CbfT [N][M].
  auto G = [&](const __bf16* A, int lda, long sA,
               const __bf16* Bt, int ldb, long sB,
               float* Cp, int ldc, long sC,
               __bf16* Cb, int ldcb, long sCb,
               __bf16* CbT, int ldcbt, long sCbt,
               const float* bias, int M, int N, int K,
               float alpha, float ident, int mode, int relu, int batch) {
    dim3 grid((unsigned)((N / 64 + 3) / 4), (unsigned)(M / 16), (unsigned)batch);
    gemm_bf16_tn<<<grid, 128, 0, stream>>>(A, lda, sA, Bt, ldb, sB,
                                           Cp, ldc, sC, Cb, ldcb, sCb,
                                           CbT, ldcbt, sCbt,
                                           bias, M, N, K, alpha, ident, mode, relu);
  };

  auto attention = [&](float* Tc, const float* ng_, const float* nb_,
                       const float* qkvw, const float* outw, const float* outb_,
                       const float* resw) {
    // x_pad = [zeros(254); LN(T)]  -> bf16
    zero_bf<<<eg((long)PADR * CDIM), 256, 0, stream>>>(XPB, (long)PADR * CDIM);
    layernorm_to_bf<<<dim3((NTOK + 3) / 4), 128, 0, stream>>>(Tc, ng_, nb_, XPB, PADR, NTOK);
    // QKV = x_pad @ Wqkv   (f32);  Wqkv^T = [1536][512]
    transpose_to_bf16<<<eg((long)CDIM * QKVD), 256, 0, stream>>>(qkvw, Wbf, CDIM, QKVD);
    G(XPB, CDIM, 0, Wbf, CDIM, 0, R2, QKVD, 0,
      nullptr, 0, 0, nullptr, 0, 0, nullptr,
      NPAD, QKVD, CDIM, 1.f, 0.f, 0, 0, 1);
    scale_q<<<eg((long)NPAD * CDIM), 256, 0, stream>>>(R2);
    f32_to_bf16_scale<<<eg((long)NPAD * QKVD), 256, 0, stream>>>(R2, R1, (long)NPAD * QKVD, 1.f);
    v_transpose_bf<<<eg((long)NHEAD * DHD * NPAD), 256, 0, stream>>>(R2, VTB);
    landmarks<<<eg((long)NHEAD * MLM * DHD), 256, 0, stream>>>(R2, QLF, KLF, QLB, KLB);

    // a1 = softmax(q @ kl^T)  : per head (NPAD x 256);  kl is already [n][k]
    G(R1, QKVD, (long)DHD, KLB, DHD, (long)MLM * DHD,
      BIG, MLM, (long)NPAD * MLM, nullptr, 0, 0, nullptr, 0, 0, nullptr,
      NPAD, MLM, DHD, 1.f, 0.f, 0, 0, NHEAD);
    softmax256<<<dim3((unsigned)(((long)NHEAD * NPAD + 7) / 8)), 256, 0, stream>>>(
        BIG, nullptr, A1B, (long)NHEAD * NPAD);

    // a2 = softmax(ql @ kl^T) : per head (256 x 256)
    G(QLB, DHD, (long)MLM * DHD, KLB, DHD, (long)MLM * DHD,
      A2F, MLM, 65536L, nullptr, 0, 0, nullptr, 0, 0, nullptr,
      MLM, MLM, DHD, 1.f, 0.f, 0, 0, NHEAD);
    softmax256<<<dim3((unsigned)((NHEAD * MLM + 7) / 8)), 256, 0, stream>>>(
        A2F, A2F, A2B, (long)NHEAD * MLM);

    // a3 = softmax(ql @ k^T)  : per head (256 x NPAD);  k tokens are [n][k]
    G(QLB, DHD, (long)MLM * DHD, R1 + CDIM, QKVD, (long)DHD,
      BIG, NPAD, (long)MLM * NPAD, nullptr, 0, 0, nullptr, 0, 0, nullptr,
      MLM, NPAD, DHD, 1.f, 0.f, 0, 0, NHEAD);
    softmax_wide<<<dim3((unsigned)(NHEAD * MLM)), 256, 0, stream>>>(BIG, A3B, NPAD);

    // Moore-Penrose pinv of a2 (6 Newton-Schulz iterations)
    pinv_stats<<<dim3(NHEAD), 256, 0, stream>>>(A2F, STAT);
    pinv_make_z<<<eg((long)NHEAD * 65536), 256, 0, stream>>>(A2F, STAT, ZF, ZB, ZBT);
    float*  zf = ZF;  __bf16* zb = ZB;  __bf16* zbt = ZBT;
    float* z2f = Z2F; __bf16* z2b = Z2B; __bf16* z2bt = Z2BT;
    for (int it = 0; it < 6; ++it) {
      // xz = x @ z  (bf16 mirror row-major: used as A below)
      G(A2B, MLM, 65536L, zbt, MLM, 65536L,
        XZF, MLM, 65536L, XZB, MLM, 65536L, nullptr, 0, 0,
        nullptr, MLM, MLM, MLM, 1.f, 0.f, 0, 0, NHEAD);
      seven_i_minus_T<<<eg((long)NHEAD * 65536), 256, 0, stream>>>(XZF, T1BT);
      // t2 = 15I - xz@t1  (transposed mirror: B-side of next)
      G(XZB, MLM, 65536L, T1BT, MLM, 65536L,
        U2F, MLM, 65536L, nullptr, 0, 0, U2BT, MLM, 65536L,
        nullptr, MLM, MLM, MLM, 1.f, 15.f, 1, 0, NHEAD);
      // t3 = 13I - xz@t2
      G(XZB, MLM, 65536L, U2BT, MLM, 65536L,
        U3F, MLM, 65536L, nullptr, 0, 0, U3BT, MLM, 65536L,
        nullptr, MLM, MLM, MLM, 1.f, 13.f, 1, 0, NHEAD);
      // z = 0.25 * z @ t3  (both row-major and transposed mirrors)
      G(zb, MLM, 65536L, U3BT, MLM, 65536L,
        z2f, MLM, 65536L, z2b, MLM, 65536L, z2bt, MLM, 65536L,
        nullptr, MLM, MLM, MLM, 0.25f, 0.f, 0, 0, NHEAD);
      { float* tf = zf; zf = z2f; z2f = tf;
        __bf16* tb = zb; zb = z2b; z2b = tb;
        __bf16* tt = zbt; zbt = z2bt; z2bt = tt; }
    }

    // G1 = a3 @ v  (256 x 64 per head);  v^T is [64][NPAD]
    G(A3B, NPAD, (long)MLM * NPAD, VTB, NPAD, (long)DHD * NPAD,
      G1F, DHD, (long)MLM * DHD, nullptr, 0, 0, G1BT, MLM, (long)DHD * MLM,
      nullptr, MLM, DHD, NPAD, 1.f, 0.f, 0, 0, NHEAD);
    // G2 = pinv(a2) @ G1;  G1^T is [64][256]
    G(zb, MLM, 65536L, G1BT, MLM, (long)DHD * MLM,
      G2F, DHD, (long)MLM * DHD, nullptr, 0, 0, G2BT, MLM, (long)DHD * MLM,
      nullptr, MLM, DHD, MLM, 1.f, 0.f, 0, 0, NHEAD);
    // OUT = a1 @ G2  -> interleaved heads into OF (NPAD x 512);  G2^T is [64][256]
    G(A1B, MLM, (long)NPAD * MLM, G2BT, MLM, (long)DHD * MLM,
      OF, CDIM, (long)DHD, nullptr, 0, 0, nullptr, 0, 0,
      nullptr, NPAD, DHD, MLM, 1.f, 0.f, 0, 0, NHEAD);

    // + depthwise residual conv over v
    res_conv<<<eg((long)NPAD * CDIM), 256, 0, stream>>>(OF, R2, resw);

    // out projection + bias, then residual into T (dropping front pad)
    f32_to_bf16_scale<<<eg((long)NPAD * CDIM), 256, 0, stream>>>(OF, OB, (long)NPAD * CDIM, 1.f);
    transpose_to_bf16<<<eg((long)CDIM * CDIM), 256, 0, stream>>>(outw, Wbf, CDIM, CDIM);
    G(OB, CDIM, 0, Wbf, CDIM, 0, R2, CDIM, 0,
      nullptr, 0, 0, nullptr, 0, 0, outb_,
      NPAD, CDIM, CDIM, 1.f, 0.f, 0, 0, 1);
    residual_add<<<eg((long)NTOK * CDIM), 256, 0, stream>>>(Tc, R2);
  };

  // ---- stem: h = relu(data @ fc1_w + b); build token buffer ----
  f32_to_bf16_scale<<<eg((long)N0 * K0), 256, 0, stream>>>(data, R1, (long)N0 * K0, 1.f);
  transpose_to_bf16<<<eg((long)K0 * CDIM), 256, 0, stream>>>(fc1_w, Wbf, K0, CDIM);
  G(R1, K0, 0, Wbf, K0, 0, R2, CDIM, 0,
    nullptr, 0, 0, nullptr, 0, 0, fc1_b,
    N0, CDIM, K0, 1.f, 0.f, 0, 1, 1);
  build_T<<<eg((long)NTOK * CDIM), 256, 0, stream>>>(R2, cls, T);

  // ---- layer 1 attention ----
  attention(T, l1_ng, l1_nb, l1_qkv, l1_ow, l1_ob, l1_rw);

  // ---- PPEG ----
  ppeg_kernel<<<eg((long)NTOK * CDIM), 256, 0, stream>>>(T, T2b, pw7, pb7, pw5, pb5, pw3, pb3);

  // ---- layer 2 attention ----
  attention(T2b, l2_ng, l2_nb, l2_qkv, l2_ow, l2_ob, l2_rw);

  // ---- head: LN(token 0) @ fc2 + softmax + argmax ----
  head_kernel<<<dim3(1), 256, 0, stream>>>(T2b, nrm_g, nrm_b, fc2_w, fc2_b, (float*)d_out);
}
